// TemporalMultiHeadAttentionBlock_71365176590701
// MI455X (gfx1250) — matrix operand order repaired and checked by reading, not compiled, not run on hardware
//
#include <hip/hip_runtime.h>

// ---- problem constants -------------------------------------------------
#define NBATCH 8
#define SEQ    512
#define NJ     25
#define IN_F   1600   // J*D
#define OUT_F  6400   // J*DH*H
#define EROWS  4096   // B*S

// ---- vector types (plain ext vectors: trivially-copyable, union-safe) --
typedef __attribute__((ext_vector_type(16))) __bf16       v16bf;
typedef __attribute__((ext_vector_type(8)))  float        v8f;
typedef __attribute__((ext_vector_type(4)))  unsigned int v4u;
typedef __attribute__((ext_vector_type(4)))  float        v4f;
typedef __attribute__((ext_vector_type(2)))  unsigned int v2u;

union FragB16 { v16bf v; v4u q[2]; };

__device__ __forceinline__ unsigned short f32_to_bf16(float f) {
  unsigned int u = __float_as_uint(f);
  u += 0x7fffu + ((u >> 16) & 1u);     // round-to-nearest-even
  return (unsigned short)(u >> 16);
}

// ---- elementwise f32 -> bf16 (vec4) ------------------------------------
__global__ __launch_bounds__(256)
void cvt_x_bf16_71365(const float* __restrict__ in, unsigned short* __restrict__ out, int n4) {
  int i = blockIdx.x * 256 + threadIdx.x;
  if (i >= n4) return;
  v4f f = ((const v4f*)in)[i];
  v2u o;
  o.x = (unsigned)f32_to_bf16(f.x) | ((unsigned)f32_to_bf16(f.y) << 16);
  o.y = (unsigned)f32_to_bf16(f.z) | ((unsigned)f32_to_bf16(f.w) << 16);
  ((v2u*)out)[i] = o;
}

// ---- LDS-tiled transpose + convert: W[rows,cols] f32 -> Wt[cols,rows] bf16
__global__ __launch_bounds__(256)
void cvt_wT_bf16_71365(const float* __restrict__ W, unsigned short* __restrict__ Wt,
                       int rows, int cols) {
  __shared__ float tile[32][33];
  const int c0 = blockIdx.x * 32, r0 = blockIdx.y * 32;
  const int tx = threadIdx.x & 31, ty = threadIdx.x >> 5;   // 32x8
#pragma unroll
  for (int i = 0; i < 32; i += 8)
    tile[ty + i][tx] = W[(long long)(r0 + ty + i) * cols + c0 + tx];
  __syncthreads();
#pragma unroll
  for (int i = 0; i < 32; i += 8)
    Wt[(long long)(c0 + ty + i) * rows + r0 + tx] = f32_to_bf16(tile[tx][ty + i]);
}

// ---- batched NT GEMM: C = A[M,K] * Bt[N,K]^T, bf16 in, f32 accumulate --
// Block = 4 waves; wave w computes rows [blk_m + 32w, +32) x cols [n0, +64)
// as 2x4 tiles of 16x16 (8 accumulators, A/B fragment reuse).
// MODE: 0 = bf16 C[M,N]; 1 = bf16 V^T per 512-row batch (Vt[b][n][t]);
//       2 = f32 scaled scores. CSCORES: skip fully-masked blocks (uniform).
// CK: clip K-loop to lower-triangular extent (attn zero-padded above diag).
template <int MODE, bool CSCORES, bool CK>
__global__ __launch_bounds__(128)
void gemm_nt_wmma_71365(const unsigned short* __restrict__ A,
                        const unsigned short* __restrict__ Bt,
                        void* __restrict__ Cv,
                        int M, int N, int K,
                        long long sA, long long sB, long long sC, float scale) {
  const long long zb = blockIdx.z;
  const int mBlk = blockIdx.y * 128;
  const int n0   = blockIdx.x * 64;

  if (CSCORES) {                  // block-uniform causal skip: SALU branch only
    if (n0 > mBlk + 127) return;
  }
  int kend = K;
  if (CK) {                       // block-uniform K clip
    const int ke = mBlk + 128;
    kend = ke < K ? ke : K;
  }

  A  += zb * sA;
  Bt += zb * sB;

  // force wave-uniform scalar so no lane-masked control flow is generated
  const int wid  = __builtin_amdgcn_readfirstlane(threadIdx.x >> 5);
  const int lane = threadIdx.x & 31;
  const int kh   = lane >> 4;     // lane half selects K sub-chunk
  const int r    = lane & 15;
  const int m0   = mBlk + wid * 32;

  // A fragment rows (16-bit A 16x32 layout: 8 bf16 at k0+kh*8, 8 at +16)
  const unsigned short* Arow0 = A + (long long)(m0 + r) * K + kh * 8;
  const unsigned short* Arow1 = Arow0 + (long long)16 * K;
  // B fragment (column-major load): Bt row n0+16t+r, 8 bf16 at k0+kh*8, 8 at +16
  const unsigned short* Brow = Bt + (long long)(n0 + r) * K + kh * 8;

  const v8f vz = {0.f, 0.f, 0.f, 0.f, 0.f, 0.f, 0.f, 0.f};
  v8f acc[2][4] = {{vz, vz, vz, vz}, {vz, vz, vz, vz}};

  for (int k0 = 0; k0 < kend; k0 += 32) {
    FragB16 a0, a1;
    a0.q[0] = *(const v4u*)(Arow0 + k0);
    a0.q[1] = *(const v4u*)(Arow0 + k0 + 16);
    a1.q[0] = *(const v4u*)(Arow1 + k0);
    a1.q[1] = *(const v4u*)(Arow1 + k0 + 16);
#pragma unroll
    for (int t = 0; t < 4; ++t) {
      const unsigned short* bp = Brow + (long long)t * 16 * K + k0;
      FragB16 b;
      b.q[0] = *(const v4u*)(bp);
      b.q[1] = *(const v4u*)(bp + 16);
      acc[0][t] = __builtin_amdgcn_wmma_f32_16x16x32_bf16(
          false, a0.v, false, b.v, (short)0, acc[0][t], false, false);
      acc[1][t] = __builtin_amdgcn_wmma_f32_16x16x32_bf16(
          false, a1.v, false, b.v, (short)0, acc[1][t], false, false);
    }
  }

  // C/D layout: VGPR i -> M = i + 8*kh (within tile), N = r (within tile)
#pragma unroll
  for (int mi = 0; mi < 2; ++mi) {
    const int gmb = m0 + mi * 16 + kh * 8;
#pragma unroll
    for (int t = 0; t < 4; ++t) {
      const int gn = n0 + t * 16 + r;
#pragma unroll
      for (int i = 0; i < 8; ++i) {
        const int gm = gmb + i;
        const float val = acc[mi][t][i];
        if (MODE == 0) {
          ((unsigned short*)Cv)[zb * sC + (long long)gm * N + gn] = f32_to_bf16(val);
        } else if (MODE == 1) {
          const int bb = gm >> 9, tt = gm & 511;    // row -> (batch, time)
          ((unsigned short*)Cv)[(long long)bb * sC + (long long)gn * 512 + tt] =
              f32_to_bf16(val);
        } else {
          ((float*)Cv)[zb * sC + (long long)gm * N + gn] = val * scale;
        }
      }
    }
  }
}

// ---- causal softmax: one wave per row, 512 cols, bf16 out (zeros masked)
__global__ __launch_bounds__(256)
void softmax_causal_71365(const float* __restrict__ scores,
                          unsigned short* __restrict__ attn) {
  const int wid = __builtin_amdgcn_readfirstlane(threadIdx.x >> 5);
  const int lane = threadIdx.x & 31;
  const int row = blockIdx.x * 8 + wid;          // 0..4095
  const int bb = row >> 9, s = row & 511;
  const float* src = scores + (long long)bb * SEQ * SEQ + (long long)s * SEQ;
  unsigned short* dst = attn + (long long)bb * SEQ * SEQ + (long long)s * SEQ;

  float vals[16];
  float m = -__builtin_inff();
#pragma unroll
  for (int i = 0; i < 16; ++i) {
    const int t = lane + i * 32;
    const float v = (t <= s) ? src[t] : -__builtin_inff();
    vals[i] = v;
    m = fmaxf(m, v);
  }
#pragma unroll
  for (int off = 16; off; off >>= 1) m = fmaxf(m, __shfl_xor(m, off, 32));

  float sum = 0.f;
#pragma unroll
  for (int i = 0; i < 16; ++i) {
    const float e = __expf(vals[i] - m);          // exp(-inf)=0 for masked
    vals[i] = e;
    sum += e;
  }
#pragma unroll
  for (int off = 16; off; off >>= 1) sum += __shfl_xor(sum, off, 32);
  const float inv = 1.f / sum;                     // t=0 always valid -> sum>0
#pragma unroll
  for (int i = 0; i < 16; ++i)
    dst[lane + i * 32] = f32_to_bf16(vals[i] * inv);
}

// ---- output projection: ctx[B*S, 6400] viewed as [102400, 256] x WbT[64,256]^T
__global__ __launch_bounds__(128)
void proj_out_wmma_71365(const unsigned short* __restrict__ ctx,
                         const unsigned short* __restrict__ wbT,
                         float* __restrict__ out) {
  const int wid = __builtin_amdgcn_readfirstlane(threadIdx.x >> 5); // n-tile
  const int lane = threadIdx.x & 31;
  const int kh = lane >> 4, r = lane & 15;
  const int m0 = blockIdx.x * 16;
  const int n0 = wid * 16;

  const int row = m0 + r;                         // row = bs*25 + j
  const int bs = row / 25;
  const int j  = row - bs * 25;
  const unsigned short* Arow = ctx + (long long)bs * OUT_F + j * 256 + kh * 8;
  const unsigned short* Brow = wbT + (long long)(n0 + r) * 256 + kh * 8;

  const v8f vz = {0.f, 0.f, 0.f, 0.f, 0.f, 0.f, 0.f, 0.f};
  v8f acc = vz;
#pragma unroll
  for (int k0 = 0; k0 < 256; k0 += 32) {
    FragB16 a, b;
    a.q[0] = *(const v4u*)(Arow + k0);
    a.q[1] = *(const v4u*)(Arow + k0 + 16);
    b.q[0] = *(const v4u*)(Brow + k0);
    b.q[1] = *(const v4u*)(Brow + k0 + 16);
    acc = __builtin_amdgcn_wmma_f32_16x16x32_bf16(
        false, a.v, false, b.v, (short)0, acc, false, false);
  }
  const int gmb = m0 + kh * 8;
#pragma unroll
  for (int i = 0; i < 8; ++i)
    out[(long long)(gmb + i) * 64 + n0 + r] = acc[i];
}

// ------------------------------------------------------------------------
extern "C" void kernel_launch(void* const* d_in, const int* in_sizes, int n_in,
                              void* d_out, int out_size, void* d_ws, size_t ws_size,
                              hipStream_t stream) {
  (void)in_sizes; (void)n_in; (void)out_size; (void)ws_size;
  const float* x  = (const float*)d_in[0];
  const float* Wq = (const float*)d_in[1];
  const float* Wk = (const float*)d_in[2];
  const float* Wv = (const float*)d_in[3];
  const float* Wb = (const float*)d_in[4];
  float* out = (float*)d_out;

  char* ws = (char*)d_ws;
  size_t off = 0;
  auto take = [&](size_t bytes) -> char* {
    char* p = ws + off;
    off = (off + bytes + 255) & ~(size_t)255;
    return p;
  };
  unsigned short* xb   = (unsigned short*)take((size_t)EROWS * IN_F * 2);
  unsigned short* wqT  = (unsigned short*)take((size_t)OUT_F * IN_F * 2);
  unsigned short* wkT  = (unsigned short*)take((size_t)OUT_F * IN_F * 2);
  unsigned short* wvT  = (unsigned short*)take((size_t)OUT_F * IN_F * 2);
  unsigned short* wbT  = (unsigned short*)take((size_t)64 * 256 * 2);
  unsigned short* Qb   = (unsigned short*)take((size_t)EROWS * OUT_F * 2);
  unsigned short* Kb   = (unsigned short*)take((size_t)EROWS * OUT_F * 2);
  unsigned short* Vt   = (unsigned short*)take((size_t)NBATCH * OUT_F * SEQ * 2);
  float*          scF  = (float*)         take((size_t)NBATCH * SEQ * SEQ * 4);
  unsigned short* attn = (unsigned short*)take((size_t)NBATCH * SEQ * SEQ * 2);
  unsigned short* ctx  = (unsigned short*)take((size_t)EROWS * OUT_F * 2);

  // 1) convert inputs to bf16 (weights transposed for NT GEMMs)
  const int n4 = (EROWS * IN_F) / 4;
  cvt_x_bf16_71365<<<dim3((n4 + 255) / 256), 256, 0, stream>>>(x, xb, n4);
  cvt_wT_bf16_71365<<<dim3(OUT_F / 32, IN_F / 32), 256, 0, stream>>>(Wq, wqT, IN_F, OUT_F);
  cvt_wT_bf16_71365<<<dim3(OUT_F / 32, IN_F / 32), 256, 0, stream>>>(Wk, wkT, IN_F, OUT_F);
  cvt_wT_bf16_71365<<<dim3(OUT_F / 32, IN_F / 32), 256, 0, stream>>>(Wv, wvT, IN_F, OUT_F);
  cvt_wT_bf16_71365<<<dim3(64 / 32, 256 / 32), 256, 0, stream>>>(Wb, wbT, 256, 64);

  // 2) Q, K (bf16 row-major), V (bf16 transposed per batch)
  gemm_nt_wmma_71365<0, false, false>
      <<<dim3(OUT_F / 64, EROWS / 128, 1), 128, 0, stream>>>(
          xb, wqT, Qb, EROWS, OUT_F, IN_F, 0, 0, 0, 1.f);
  gemm_nt_wmma_71365<0, false, false>
      <<<dim3(OUT_F / 64, EROWS / 128, 1), 128, 0, stream>>>(
          xb, wkT, Kb, EROWS, OUT_F, IN_F, 0, 0, 0, 1.f);
  gemm_nt_wmma_71365<1, false, false>
      <<<dim3(OUT_F / 64, EROWS / 128, 1), 128, 0, stream>>>(
          xb, wvT, Vt, EROWS, OUT_F, IN_F, 0, 0, (long long)OUT_F * SEQ, 1.f);

  // 3) scores = Q K^T * d^-0.5 (f32), fully-masked blocks skipped
  gemm_nt_wmma_71365<2, true, false>
      <<<dim3(SEQ / 64, SEQ / 128, NBATCH), 128, 0, stream>>>(
          Qb, Kb, scF, SEQ, SEQ, OUT_F,
          (long long)SEQ * OUT_F, (long long)SEQ * OUT_F, (long long)SEQ * SEQ,
          0.125f);

  // 4) causal softmax -> bf16 attn (zeros above diagonal)
  softmax_causal_71365<<<dim3(EROWS / 8), 256, 0, stream>>>(scF, attn);

  // 5) ctx = attn @ V  (NT against V^T), K-loop clipped to lower triangle
  gemm_nt_wmma_71365<0, false, true>
      <<<dim3(OUT_F / 64, SEQ / 128, NBATCH), 128, 0, stream>>>(
          attn, Vt, ctx, SEQ, OUT_F, SEQ,
          (long long)SEQ * SEQ, (long long)OUT_F * SEQ, (long long)SEQ * OUT_F,
          1.f);

  // 6) output projection -> f32 out [B,S,J,64]
  proj_out_wmma_71365<<<dim3((EROWS * NJ) / 16), 128, 0, stream>>>(ctx, wbT, out);
}
